// LSTMModel2_43911745634898
// MI455X (gfx1250) — matrix operand order repaired and checked by reading, not compiled, not run on hardware
//
#include <hip/hip_runtime.h>
#include <stdint.h>

// Problem constants (from the reference)
#define NB    256   // batch
#define SEQL  2048  // timesteps
#define DIN   8     // input features
#define H1D   256   // LSTM hidden
#define H2D   128   // MLP hidden
#define G4    1024  // 4*H1 gate width

typedef _Float16 v16h __attribute__((ext_vector_type(16)));
typedef float    v8f  __attribute__((ext_vector_type(8)));
typedef uint32_t v4u  __attribute__((ext_vector_type(4)));

union FragU { v16h h; v4u u[2]; };

// A-matrix fragment (16x32 f16) in contiguous k order (not the ISA table order):
// lane<16: K0..15 contiguous ; lane>=16: K16..31 contiguous (matches packed B)
__device__ __forceinline__ v16h load_a_frag(const _Float16* rowBase, int kbase, int half) {
    const _Float16* p = rowBase + kbase + 16 * half;
    FragU u;
    u.u[0] = *(const v4u*)(p);        // 8 f16 = K 16h+0..7
    u.u[1] = *(const v4u*)(p + 8);    // 8 f16 = K 16h+8..15
    return u.h;
}

// B-matrix fragment from pre-packed memory: 32 lanes x 16 f16, lane-contiguous 32B
__device__ __forceinline__ v16h load_b_frag(const _Float16* pack, int fragIdx, int lane) {
    const _Float16* p = pack + ((size_t)fragIdx * 32 + lane) * 16;
    FragU u;
    u.u[0] = *(const v4u*)(p);
    u.u[1] = *(const v4u*)(p + 8);
    return u.h;
}

__device__ __forceinline__ v8f wmma16(v16h a, v16h b, v8f c) {
    return __builtin_amdgcn_wmma_f32_16x16x32_f16(false, a, false, b, (short)0, c, false, false);
}

// Fast gate nonlinearities: v_exp_f32 + v_rcp_f32 (instead of libm tanhf)
__device__ __forceinline__ float fast_sigmoid(float x) {
    return __builtin_amdgcn_rcpf(1.f + __expf(-x));
}
__device__ __forceinline__ float fast_tanh(float x) {
    return 2.f * __builtin_amdgcn_rcpf(1.f + __expf(-2.f * x)) - 1.f;
}

// ---------------------------------------------------------------------------
// Kernel A: pack weights to f16 in WMMA B-fragment layout.
// B-frag layout: lane l holds column N = tile*16 + (l&15); in-lane f16 element
// e (0..15) holds K = kt*32 + 16*(l>>4) + e.
// ---------------------------------------------------------------------------
__global__ void pack_weights(const float* __restrict__ Wih,   // [1024][9]
                             const float* __restrict__ Whh,   // [1024][256]
                             const float* __restrict__ W1,    // [128][256]
                             const float* __restrict__ bih,   // [1024]
                             const float* __restrict__ bhh,   // [1024]
                             _Float16* __restrict__ pWhh,     // 64 tiles * 8 kt * 512 f16
                             _Float16* __restrict__ pWih,     // 64 tiles * 512 f16 (K=32 padded)
                             _Float16* __restrict__ pW1,      // 8 tiles * 8 kt * 512 f16
                             float* __restrict__ bsum) {      // [1024]
    const int stride = gridDim.x * blockDim.x;
    const int tid0 = blockIdx.x * blockDim.x + threadIdx.x;

    for (int idx = tid0; idx < 64 * 8 * 512; idx += stride) {
        int e = idx & 15, lane = (idx >> 4) & 31, frag = idx >> 9;
        int kt = frag & 7, tile = frag >> 3;
        int n = tile * 16 + (lane & 15);
        int k = kt * 32 + (lane >> 4) * 16 + e;
        pWhh[idx] = (_Float16)Whh[n * 256 + k];
    }
    for (int idx = tid0; idx < 8 * 8 * 512; idx += stride) {
        int e = idx & 15, lane = (idx >> 4) & 31, frag = idx >> 9;
        int kt = frag & 7, tile = frag >> 3;
        int n = tile * 16 + (lane & 15);          // < 128
        int k = kt * 32 + (lane >> 4) * 16 + e;   // < 256
        pW1[idx] = (_Float16)W1[n * 256 + k];
    }
    for (int idx = tid0; idx < 64 * 512; idx += stride) {
        int e = idx & 15, lane = (idx >> 4) & 31, tile = idx >> 9;
        int n = tile * 16 + (lane & 15);
        int k = (lane >> 4) * 16 + e;             // < 32 (padded K)
        pWih[idx] = (_Float16)((k < DIN + 1) ? Wih[n * (DIN + 1) + k] : 0.f);
    }
    for (int idx = tid0; idx < G4; idx += stride) bsum[idx] = bih[idx] + bhh[idx];
}

// ---------------------------------------------------------------------------
// Kernel B: persistent LSTM. 16 WGs x 256 threads; each WG owns 16 batch rows
// and runs all 2048 timesteps with 4 intra-WG barriers per step.
// ---------------------------------------------------------------------------
__global__ __launch_bounds__(256, 1)
void lstm_persistent(const float* __restrict__ x,       // [N][L][8]
                     const float* __restrict__ b1,      // [128]
                     const float* __restrict__ W2,      // [128]
                     const float* __restrict__ b2,      // [1]
                     const _Float16* __restrict__ pWhh,
                     const _Float16* __restrict__ pWih,
                     const _Float16* __restrict__ pW1,
                     const float* __restrict__ bsum,
                     float* __restrict__ outBuf,        // [N*L]
                     float* __restrict__ cellBuf) {     // [N*L*H1]
    __shared__ _Float16 sH[16 * H1D];     // current h (f16) for A-frags
    __shared__ _Float16 sInp[16 * 32];    // [x_t(8) | prev(1) | zeros] K=32 padded
    __shared__ _Float16 sGates[16 * G4];  // pre-activation gates (f16)
    __shared__ float    sA1[16 * H2D];    // relu(h@W1^T + b1)
    __shared__ float    sBias[G4];        // b_ih + b_hh
    __shared__ float    sB1[H2D];
    __shared__ float    sW2[H2D];

    const int tid   = threadIdx.x;
    const int wave  = tid >> 5;
    const int lane  = tid & 31;
    const int half  = lane >> 4;
    const int col16 = lane & 15;
    const int mA    = lane & 15;          // A-row this lane loads
    const int m0    = blockIdx.x * 16;    // batch row base

    // cell state registers: this thread owns (mC, j0 + 16k) for k=0..15
    const int mC = tid >> 4;
    const int j0 = tid & 15;
    float creg[16];
#pragma unroll
    for (int k = 0; k < 16; ++k) creg[k] = 0.f;

    // init LDS: zero h and padded input, cache biases/W2, load x at t=0
    for (int i = tid; i < 16 * 32; i += 256) sInp[i] = (_Float16)0.f;
    for (int i = tid; i < 16 * H1D; i += 256) sH[i] = (_Float16)0.f;
    for (int i = tid; i < G4; i += 256) sBias[i] = bsum[i];
    if (tid < H2D) { sB1[tid] = b1[tid]; sW2[tid] = W2[tid]; }
    if (tid < 128) {
        int m = tid >> 3, f = tid & 7;
        sInp[m * 32 + f] = (_Float16)x[((size_t)(m0 + m) * SEQL + 0) * DIN + f];
    }
    const float b2v = b2[0];
    __syncthreads();

    for (int t = 0; t < SEQL; ++t) {
        // ---- Phase 1: gates = inp@Wih^T + h@Whh^T + (b_ih+b_hh) via WMMA ----
        {
            v16h aI = load_a_frag(sInp + mA * 32, 0, half);
#pragma unroll 1
            for (int ip = 0; ip < 4; ++ip) {      // pairs of gate n-tiles
                const int t0 = wave * 8 + 2 * ip;
                const float bias0 = sBias[t0 * 16 + col16];
                const float bias1 = sBias[(t0 + 1) * 16 + col16];
                v8f acc0, acc1;
#pragma unroll
                for (int r = 0; r < 8; ++r) { acc0[r] = bias0; acc1[r] = bias1; }
                // padded-input contribution (K=32: 8 x-features + prev + zeros)
                acc0 = wmma16(aI, load_b_frag(pWih, t0, lane), acc0);
                acc1 = wmma16(aI, load_b_frag(pWih, t0 + 1, lane), acc1);
#pragma unroll 2
                for (int kt = 0; kt < 8; ++kt) {  // K = 256 over h
                    v16h aH = load_a_frag(sH + mA * H1D, kt * 32, half);
                    acc0 = wmma16(aH, load_b_frag(pWhh, t0 * 8 + kt, lane), acc0);
                    acc1 = wmma16(aH, load_b_frag(pWhh, (t0 + 1) * 8 + kt, lane), acc1);
                }
                // C/D layout: element (M = r + 8*half, N = col16)
#pragma unroll
                for (int r = 0; r < 8; ++r) {
                    sGates[(r + 8 * half) * G4 + t0 * 16 + col16]       = (_Float16)acc0[r];
                    sGates[(r + 8 * half) * G4 + (t0 + 1) * 16 + col16] = (_Float16)acc1[r];
                }
            }
        }
        __syncthreads();

        // ---- Phase 2: LSTM cell update (c in registers) ----
        {
            float* cOut = cellBuf + ((size_t)(m0 + mC) * SEQL + t) * H1D;
#pragma unroll 4
            for (int k = 0; k < 16; ++k) {
                int jj = j0 + 16 * k;
                float gi = (float)sGates[mC * G4 + jj];
                float gf = (float)sGates[mC * G4 + 256 + jj];
                float gg = (float)sGates[mC * G4 + 512 + jj];
                float go = (float)sGates[mC * G4 + 768 + jj];
                float ii = fast_sigmoid(gi);
                float ff = fast_sigmoid(gf);
                float gv = fast_tanh(gg);
                float oo = fast_sigmoid(go);
                float c  = ff * creg[k] + ii * gv;
                creg[k]  = c;
                float h  = oo * fast_tanh(c);
                sH[mC * H1D + jj] = (_Float16)h;
                cOut[jj] = c;
            }
        }
        __syncthreads();

        // ---- Phase 3: a1 = relu(h @ W1^T + b1) via WMMA (wave == n-tile) ----
        {
            float bias = sB1[wave * 16 + col16];
            v8f a;
#pragma unroll
            for (int r = 0; r < 8; ++r) a[r] = bias;
#pragma unroll 2
            for (int kt = 0; kt < 8; ++kt) {
                v16h aH = load_a_frag(sH + mA * H1D, kt * 32, half);
                v16h bW = load_b_frag(pW1, wave * 8 + kt, lane);
                a = wmma16(aH, bW, a);
            }
#pragma unroll
            for (int r = 0; r < 8; ++r) {
                float v = a[r];
                sA1[(r + 8 * half) * H2D + wave * 16 + col16] = v > 0.f ? v : 0.f;
            }
        }
        __syncthreads();

        // ---- Phase 4: scalar head via 16-lane parallel dot + shuffle tree ----
        {
            float partial = 0.f;
#pragma unroll
            for (int k = 0; k < 8; ++k) {
                int j = j0 + 16 * k;
                partial += sA1[mC * H2D + j] * sW2[j];
            }
            // reduce within each 16-lane group (xor masks stay inside the group)
#pragma unroll
            for (int s = 8; s >= 1; s >>= 1)
                partial += __shfl_xor(partial, s, 32);
            if (j0 == 0) {
                float sOut = partial + b2v;
                outBuf[(size_t)(m0 + mC) * SEQL + t] = sOut;
                sInp[mC * 32 + 8] = (_Float16)sOut;   // prev_output column
            }
            if (tid < 128 && t + 1 < SEQL) {
                int m = tid >> 3, f = tid & 7;
                sInp[m * 32 + f] = (_Float16)x[((size_t)(m0 + m) * SEQL + (t + 1)) * DIN + f];
            }
        }
        __syncthreads();
    }
}

// ---------------------------------------------------------------------------
extern "C" void kernel_launch(void* const* d_in, const int* in_sizes, int n_in,
                              void* d_out, int out_size, void* d_ws, size_t ws_size,
                              hipStream_t stream) {
    (void)in_sizes; (void)n_in; (void)out_size;

    const float* x   = (const float*)d_in[0];
    const float* Wih = (const float*)d_in[1];
    const float* Whh = (const float*)d_in[2];
    const float* bih = (const float*)d_in[3];
    const float* bhh = (const float*)d_in[4];
    const float* W1  = (const float*)d_in[5];
    const float* b1  = (const float*)d_in[6];
    const float* W2  = (const float*)d_in[7];
    const float* b2  = (const float*)d_in[8];

    // workspace layout (all 16B aligned)
    const size_t szWhh = (size_t)64 * 8 * 512 * sizeof(_Float16); // 512 KB
    const size_t szWih = (size_t)64 * 512 * sizeof(_Float16);     //  64 KB
    const size_t szW1  = (size_t)8 * 8 * 512 * sizeof(_Float16);  //  64 KB
    const size_t szB   = (size_t)G4 * sizeof(float);              //   4 KB
    if (ws_size < szWhh + szWih + szW1 + szB) return;

    char* ws = (char*)d_ws;
    _Float16* pWhh = (_Float16*)(ws);
    _Float16* pWih = (_Float16*)(ws + szWhh);
    _Float16* pW1  = (_Float16*)(ws + szWhh + szWih);
    float*    bsum = (float*)(ws + szWhh + szWih + szW1);

    float* outBuf  = (float*)d_out;                         // [N*L*OUT], OUT=1
    float* cellBuf = (float*)d_out + (size_t)NB * SEQL;     // [N*L*H1]

    pack_weights<<<dim3(256), dim3(256), 0, stream>>>(
        Wih, Whh, W1, bih, bhh, pWhh, pWih, pW1, bsum);
    lstm_persistent<<<dim3(16), dim3(256), 0, stream>>>(
        x, b1, W2, b2, pWhh, pWih, pW1, bsum, outBuf, cellBuf);
}
